// CompositionModel_16741782520617
// MI455X (gfx1250) — matrix operand order repaired and checked by reading, not compiled, not run on hardware
//
#include <hip/hip_runtime.h>

#define N_CELLS 500000
#define DX 128
#define DZ 32
#define DIN 160
#define HDIM 256
#define DOUT 16
#define MT 128          // cells per workgroup (8 waves x 16 rows)
#define SX 168          // xp LDS row stride in halfs (padded, 16B-aligned rows)
#define SH 264          // h1/h2 LDS row stride in halfs (padded, 16B-aligned rows)

typedef _Float16 half8  __attribute__((ext_vector_type(8)));
typedef _Float16 half16 __attribute__((ext_vector_type(16)));
typedef float    f32x8  __attribute__((ext_vector_type(8)));

__device__ __forceinline__ half16 cat16(half8 lo, half8 hi) {
  return __builtin_shufflevector(lo, hi, 0,1,2,3,4,5,6,7,8,9,10,11,12,13,14,15);
}

// A fragment (16xK tile, f16): lane = M row; lanes 0-15: halfs K0..7,K16..23;
// lanes 16-31: halfs K8..15,K24..31  (ISA 7.12.2 16-bit A-matrix table)
__device__ __forceinline__ half16 load_A(const _Float16* base, int stride,
                                         int rb, int k0, int lane) {
  const _Float16* p = base + (rb + (lane & 15)) * stride + k0 + ((lane >> 4) << 3);
  return cat16(*(const half8*)p, *(const half8*)(p + 16));
}

// B fragment (32x16 KxN tile, f16) from row-major W^T[N][K]: lane = N col;
// lanes 0-15: halfs K0..7,K16..23; lanes 16-31: halfs K8..15,K24..31 (matches A)
__device__ __forceinline__ half16 load_B(const _Float16* Wt, int K,
                                         int n0, int k0, int lane) {
  const _Float16* p = Wt + (n0 + (lane & 15)) * K + k0 + ((lane >> 4) << 3);
  return cat16(*(const half8*)p, *(const half8*)(p + 16));
}

__device__ __forceinline__ f32x8 wmma16(half16 a, half16 b, f32x8 c) {
  return __builtin_amdgcn_wmma_f32_16x16x32_f16(false, a, false, b,
                                                (short)0, c, false, false);
}

// --- weight transpose/convert + accumulator zeroing ---
__global__ void prep_kernel(const float* __restrict__ W1, const float* __restrict__ W2,
                            const float* __restrict__ W3,
                            _Float16* __restrict__ W1T, _Float16* __restrict__ W2T,
                            _Float16* __restrict__ W3T,
                            float* __restrict__ sums, float* __restrict__ counts) {
  int i = blockIdx.x * 256 + threadIdx.x;           // up to 65536
  if (i < HDIM * DIN)  { int n = i / DIN, k = i % DIN;  W1T[n * DIN  + k] = (_Float16)W1[k * HDIM + n]; }
  if (i < HDIM * HDIM) { int n = i >> 8,  k = i & 255;  W2T[n * HDIM + k] = (_Float16)W2[k * HDIM + n]; }
  if (i < DOUT * HDIM) { int k = i >> 4,  n = i & 15;   W3T[n * HDIM + k] = (_Float16)W3[k * DOUT + n]; }
  if (i < 512 * DOUT)  sums[i] = 0.0f;
  if (i < 512)         counts[i] = 0.0f;
}

// --- fused 3-layer MLP, all matmuls on v_wmma_f32_16x16x32_f16 ---
__global__ __launch_bounds__(256, 1)
void mlp_fused(const float* __restrict__ X, const float* __restrict__ Z,
               const float* __restrict__ b1, const float* __restrict__ b2,
               const float* __restrict__ b3,
               const int* __restrict__ c2b, const int* __restrict__ sidx,
               const _Float16* __restrict__ W1T, const _Float16* __restrict__ W2T,
               const _Float16* __restrict__ W3T,
               float* __restrict__ sums, float* __restrict__ counts)
{
  extern __shared__ _Float16 smem[];
  _Float16* region0 = smem;                 // xp (stride SX), later h2 (stride SH)
  _Float16* h1      = smem + MT * SH;       // stride SH
  int* segArr = (int*)(smem + 2 * MT * SH); // MT ints

  const int tid    = threadIdx.x;
  const int lane   = tid & 31;
  const int wave   = tid >> 5;
  const int wgBase = blockIdx.x * MT;

  // ---- stage Xp = [log1p(X) | Z[c2b]] as f16; record segments + counts ----
  {
    const int r  = tid >> 1;         // cell row 0..127 (two threads per row)
    const int hf = tid & 1;
    const int cell  = wgBase + r;
    const bool valid = cell < N_CELLS;
    const int b = valid ? c2b[cell] : 0;
    if (hf == 0) {
      const int s = valid ? sidx[b] : 0;
      segArr[r] = s;
      if (valid) atomicAdd(&counts[s], 1.0f);
    }
    _Float16* xr = region0 + r * SX;
    if (valid) {
      const float* xs = X + (long long)cell * DX + hf * 64;
      #pragma unroll
      for (int j = 0; j < 64; j += 4) {
        float4 v = *(const float4*)(xs + j);
        xr[hf * 64 + j + 0] = (_Float16)__logf(1.0f + v.x);
        xr[hf * 64 + j + 1] = (_Float16)__logf(1.0f + v.y);
        xr[hf * 64 + j + 2] = (_Float16)__logf(1.0f + v.z);
        xr[hf * 64 + j + 3] = (_Float16)__logf(1.0f + v.w);
      }
      if (hf) {
        const float* zs = Z + b * DZ;
        #pragma unroll
        for (int j = 0; j < DZ; j += 4) {
          float4 v = *(const float4*)(zs + j);
          xr[DX + j + 0] = (_Float16)v.x;
          xr[DX + j + 1] = (_Float16)v.y;
          xr[DX + j + 2] = (_Float16)v.z;
          xr[DX + j + 3] = (_Float16)v.w;
        }
      }
    } else {
      for (int j = 0; j < 64; ++j) xr[hf * 64 + j] = (_Float16)0.0f;
      if (hf) for (int j = 0; j < DZ; ++j) xr[DX + j] = (_Float16)0.0f;
    }
  }
  __syncthreads();

  // ---- layer 1: h1 = relu(Xp @ W1 + b1), K=160 (5 chunks), 16 N-tiles ----
  #pragma unroll
  for (int tt = 0; tt < 2; ++tt) {
    const int t = wave + tt * 8;
    half16 Bf0 = load_B(W1T, DIN, t * 16, 0,   lane);
    half16 Bf1 = load_B(W1T, DIN, t * 16, 32,  lane);
    half16 Bf2 = load_B(W1T, DIN, t * 16, 64,  lane);
    half16 Bf3 = load_B(W1T, DIN, t * 16, 96,  lane);
    half16 Bf4 = load_B(W1T, DIN, t * 16, 128, lane);
    const float bv = b1[t * 16 + (lane & 15)];
    const int m0  = (lane >> 4) << 3;
    const int col = t * 16 + (lane & 15);
    for (int rb = 0; rb < MT; rb += 16) {
      f32x8 acc = {};
      acc = wmma16(load_A(region0, SX, rb, 0,   lane), Bf0, acc);
      acc = wmma16(load_A(region0, SX, rb, 32,  lane), Bf1, acc);
      acc = wmma16(load_A(region0, SX, rb, 64,  lane), Bf2, acc);
      acc = wmma16(load_A(region0, SX, rb, 96,  lane), Bf3, acc);
      acc = wmma16(load_A(region0, SX, rb, 128, lane), Bf4, acc);
      #pragma unroll
      for (int v = 0; v < 8; ++v)
        h1[(rb + m0 + v) * SH + col] = (_Float16)fmaxf(acc[v] + bv, 0.0f);
    }
  }
  __syncthreads();   // all xp reads done -> region0 reusable as h2

  // ---- layer 2: h2 = relu(h1 @ W2 + b2), K=256 (8 chunks), 16 N-tiles ----
  #pragma unroll
  for (int tt = 0; tt < 2; ++tt) {
    const int t = wave + tt * 8;
    half16 Bf[8];
    #pragma unroll
    for (int kc = 0; kc < 8; ++kc) Bf[kc] = load_B(W2T, HDIM, t * 16, kc * 32, lane);
    const float bv = b2[t * 16 + (lane & 15)];
    const int m0  = (lane >> 4) << 3;
    const int col = t * 16 + (lane & 15);
    for (int rb = 0; rb < MT; rb += 16) {
      f32x8 acc = {};
      #pragma unroll
      for (int kc = 0; kc < 8; ++kc)
        acc = wmma16(load_A(h1, SH, rb, kc * 32, lane), Bf[kc], acc);
      #pragma unroll
      for (int v = 0; v < 8; ++v)
        region0[(rb + m0 + v) * SH + col] = (_Float16)fmaxf(acc[v] + bv, 0.0f);
    }
  }
  __syncthreads();

  // ---- layer 3: out = h2 @ W3 + b3 (16 cols), one row-tile per wave,
  //      then atomic segment-sum scatter ----
  {
    const int rb = wave * 16;
    f32x8 acc = {};
    #pragma unroll
    for (int kc = 0; kc < 8; ++kc) {
      half16 bf = load_B(W3T, HDIM, 0, kc * 32, lane);
      acc = wmma16(load_A(region0, SH, rb, kc * 32, lane), bf, acc);
    }
    const float bv = b3[lane & 15];
    const int m0 = (lane >> 4) << 3;
    const int n  = lane & 15;
    #pragma unroll
    for (int v = 0; v < 8; ++v) {
      const int r = rb + m0 + v;
      const int cell = wgBase + r;
      if (cell < N_CELLS)
        atomicAdd(&sums[segArr[r] * DOUT + n], acc[v] + bv);
    }
  }
}

// --- Y = sums / max(counts, 1) ---
__global__ void finalize_kernel(const float* __restrict__ sums,
                                const float* __restrict__ counts,
                                float* __restrict__ out) {
  int i = blockIdx.x * 256 + threadIdx.x;
  if (i < 512 * DOUT) {
    float c = counts[i >> 4];
    out[i] = sums[i] / fmaxf(c, 1.0f);
  }
}

extern "C" void kernel_launch(void* const* d_in, const int* in_sizes, int n_in,
                              void* d_out, int out_size, void* d_ws, size_t ws_size,
                              hipStream_t stream) {
  const float* X  = (const float*)d_in[0];
  const float* Z  = (const float*)d_in[1];
  const float* W1 = (const float*)d_in[2];
  const float* b1 = (const float*)d_in[3];
  const float* W2 = (const float*)d_in[4];
  const float* b2 = (const float*)d_in[5];
  const float* W3 = (const float*)d_in[6];
  const float* b3 = (const float*)d_in[7];
  const int* c2b  = (const int*)d_in[8];
  const int* sidx = (const int*)d_in[9];
  float* out = (float*)d_out;

  char* ws = (char*)d_ws;
  float*    sums   = (float*)(ws + 0);          // 512*16*4   = 32768 B
  float*    counts = (float*)(ws + 32768);      // 512*4      =  2048 B
  _Float16* W1T    = (_Float16*)(ws + 36864);   // 256*160*2  = 81920 B
  _Float16* W2T    = (_Float16*)(ws + 118784);  // 256*256*2  = 131072 B
  _Float16* W3T    = (_Float16*)(ws + 249856);  // 16*256*2   =  8192 B

  prep_kernel<<<256, 256, 0, stream>>>(W1, W2, W3, W1T, W2T, W3T, sums, counts);

  const int grid = (N_CELLS + MT - 1) / MT;                      // 3907
  const size_t lds = (size_t)2 * MT * SH * sizeof(_Float16)     // xp/h2 + h1
                   + (size_t)MT * sizeof(int);                  // segArr  = 135680 B
  mlp_fused<<<grid, 256, lds, stream>>>(X, Z, b1, b2, b3, c2b, sidx,
                                        W1T, W2T, W3T, sums, counts);

  finalize_kernel<<<32, 256, 0, stream>>>(sums, counts, out);
}
